// DMFC_block_26560077759177
// MI455X (gfx1250) — matrix operand order repaired and checked by reading, not compiled, not run on hardware
//
#include <hip/hip_runtime.h>

// ---------------------------------------------------------------------------
// DMFC block for MI455X (gfx1250, wave32). Heavy contractions use
// v_wmma_f32_16x16x32_f16 (f16 in, f32 accumulate). GEMM and attention both
// stage f32->f16 tiles in LDS with coalesced float4 loads; WMMA fragments are
// read back as ds_load_b128 pairs. Attention is fused (online softmax).
// ---------------------------------------------------------------------------

typedef __attribute__((ext_vector_type(16))) _Float16 v16h;
typedef __attribute__((ext_vector_type(8)))  _Float16 h8;
typedef __attribute__((ext_vector_type(8)))  float    v8f;
typedef __attribute__((ext_vector_type(4)))  float    v4f;

constexpr int kB  = 4;      // batch
constexpr int kC  = 128;    // channels
constexpr int kH  = 4;      // heads
constexpr int kHD = 32;     // head dim
constexpr int kMS = 512;    // sampled points
constexpr int kN  = 8192;   // correspondences

// ---------------- WMMA fragment layout helpers (per CDNA5 ISA 7.12.2) ------
// 16-bit A (16x32, MxK): lane 0-15 -> M=lane, K base 0; lane 16-31 -> K base 8.
// element e (0..15): k = kbase + (e<8 ? e : e+8)
__device__ __forceinline__ int a_k_of_e(int lane, int e) {
  int kbase = (lane < 16) ? 0 : 8;
  return kbase + (e < 8 ? e : e + 8);
}
// 16-bit B (32x16, KxN): lane = N (mod 16); lanes 0-15 hold K=0..7,16..23, lanes
// 16-31 hold K=8..15,24..31 (same k order as A).

__device__ __forceinline__ v16h combine16(h8 lo, h8 hi) {
  return __builtin_shufflevector(lo, hi, 0, 1, 2, 3, 4, 5, 6, 7, 8, 9, 10, 11,
                                 12, 13, 14, 15);
}

// ---------------- Generic batched WMMA GEMM --------------------------------
// D[m,n] = sum_k opA(A)[m,k] * opB(B)[k,n] (+ bias[m]) (+ D if ACC)
// Block tile 64x64, K-step 32, 4 waves; each wave computes 16(m) x 64(n).
// Requires M, N multiples of 64 and K multiple of 32 (true for all calls).
constexpr int BM = 64, BN = 64, BK = 32;
constexpr int LP = BK + 8;  // padded LDS row stride (halves) to spread banks

template <bool TA, bool TB, bool TD, bool ACC>
__global__ __launch_bounds__(128) void gemm_wmma_kernel(
    const float* __restrict__ A, int lda, size_t sA, const float* __restrict__ B,
    int ldb, size_t sB, float* __restrict__ D, int ldd, size_t sD,
    const float* __restrict__ bias, int M, int N, int K) {
  __shared__ _Float16 As[BM * LP];  // [m][k]
  __shared__ _Float16 Bs[BN * LP];  // [n][k] (transposed stage)

  const int lane = threadIdx.x;
  const int wave = threadIdx.y;
  const int tid = wave * 32 + lane;
  const int m_blk = blockIdx.y * BM;
  const int n_blk = blockIdx.x * BN;
  if (m_blk >= M || n_blk >= N) return;
  const int bz = blockIdx.z;
  A += (size_t)bz * sA;
  B += (size_t)bz * sB;
  D += (size_t)bz * sD;

  const int nn = lane & 15;            // C/D: lane = n
  const int mb = (lane < 16) ? 0 : 8;  // C/D: vgpr r -> m = mb + r
  const int kbA = (lane < 16) ? 0 : 8;
  const int kbB = (lane < 16) ? 0 : 8;
  const int mw = wave * 16;  // wave's m-subtile inside block tile

  v8f acc[4];
#pragma unroll
  for (int t = 0; t < 4; ++t) {
#pragma unroll
    for (int r = 0; r < 8; ++r) {
      if (ACC) {
        int gm = m_blk + mw + mb + r, gn = n_blk + t * 16 + nn;
        acc[t][r] = TD ? D[(size_t)gn * ldd + gm] : D[(size_t)gm * ldd + gn];
      } else {
        acc[t][r] = 0.0f;
      }
    }
  }

  for (int k0 = 0; k0 < K; k0 += BK) {
    __syncthreads();  // previous iteration's LDS reads done
    // ---- stage A tile (64 x 32) into As[m][k], f32 -> f16 ----
#pragma unroll
    for (int rep = 0; rep < 4; ++rep) {
      int idx = rep * 128 + tid;  // 0..511 quads
      int m = idx >> 3, kq = (idx & 7) * 4;
      if (!TA) {
        v4f v = *(const v4f*)(A + (size_t)(m_blk + m) * lda + k0 + kq);
#pragma unroll
        for (int j = 0; j < 4; ++j) As[m * LP + kq + j] = (_Float16)v[j];
      } else {
#pragma unroll
        for (int j = 0; j < 4; ++j)
          As[m * LP + kq + j] = (_Float16)A[(size_t)(k0 + kq + j) * lda + m_blk + m];
      }
    }
    // ---- stage B tile (32 x 64) into Bs[n][k] ----
#pragma unroll
    for (int rep = 0; rep < 4; ++rep) {
      int idx = rep * 128 + tid;
      if (!TB) {  // B row-major [K,N]: coalesced along n
        int k = idx >> 4, nq = (idx & 15) * 4;
        v4f v = *(const v4f*)(B + (size_t)(k0 + k) * ldb + n_blk + nq);
#pragma unroll
        for (int j = 0; j < 4; ++j) Bs[(nq + j) * LP + k] = (_Float16)v[j];
      } else {  // B stored [N,K]: coalesced along k
        int n = idx >> 3, kq = (idx & 7) * 4;
        v4f v = *(const v4f*)(B + (size_t)(n_blk + n) * ldb + k0 + kq);
#pragma unroll
        for (int j = 0; j < 4; ++j) Bs[n * LP + kq + j] = (_Float16)v[j];
      }
    }
    if (k0 + BK < K) {  // hint next slab toward the caches (global_prefetch_b8)
      __builtin_prefetch(A + (TA ? (size_t)(k0 + BK) * lda + m_blk + tid
                                 : (size_t)(m_blk + (tid >> 1)) * lda + k0 + BK), 0, 1);
      __builtin_prefetch(B + (TB ? (size_t)(n_blk + (tid >> 1)) * ldb + k0 + BK
                                 : (size_t)(k0 + BK + (tid >> 2)) * ldb + n_blk), 0, 1);
    }
    __syncthreads();

    // ---- A fragment: two b128 LDS loads ----
    const _Float16* ap = As + (size_t)(mw + nn) * LP + kbA;
    v16h afrag = combine16(*(const h8*)(ap), *(const h8*)(ap + 16));
    // ---- 4 B fragments, 4 WMMAs (A reused) ----
#pragma unroll
    for (int t = 0; t < 4; ++t) {
      const _Float16* bp = Bs + (size_t)(t * 16 + nn) * LP + kbB;
      v16h bfrag = combine16(*(const h8*)(bp), *(const h8*)(bp + 16));
      acc[t] = __builtin_amdgcn_wmma_f32_16x16x32_f16(false, afrag, false, bfrag,
                                                      (short)0, acc[t], false,
                                                      false);
    }
  }

#pragma unroll
  for (int t = 0; t < 4; ++t) {
#pragma unroll
    for (int r = 0; r < 8; ++r) {
      int gm = m_blk + mw + mb + r, gn = n_blk + t * 16 + nn;
      float v = acc[t][r];
      if (bias) v += bias[gm];
      if (TD)
        D[(size_t)gn * ldd + gm] = v;
      else
        D[(size_t)gm * ldd + gn] = v;
    }
  }
}

static void launch_gemm(hipStream_t st, const float* A, int lda, size_t sA,
                        const float* B, int ldb, size_t sB, float* D, int ldd,
                        size_t sD, const float* bias, int M, int N, int K,
                        int batch, bool ta, bool tb, bool td, bool acc) {
  dim3 grid((unsigned)(N / BN), (unsigned)(M / BM), (unsigned)batch);
  dim3 block(32, 4, 1);
#define GEMM_CASE(TA, TB, TD, ACC)                                            \
  gemm_wmma_kernel<TA, TB, TD, ACC><<<grid, block, 0, st>>>(                  \
      A, lda, sA, B, ldb, sB, D, ldd, sD, bias, M, N, K)
  if (!ta && !tb && !td && !acc)      GEMM_CASE(false, false, false, false);
  else if (!ta && !tb && !td && acc)  GEMM_CASE(false, false, false, true);
  else if (!ta && tb && !td && !acc)  GEMM_CASE(false, true, false, false);
  else if (ta && !tb && !td && !acc)  GEMM_CASE(true, false, false, false);
  else if (!ta && !tb && td && !acc)  GEMM_CASE(false, false, true, false);
#undef GEMM_CASE
}

// ---------------- Fused attention (flash-style, WMMA, LDS-staged K/V) ------
// qf/kf/vf: [B, C, N*] f32 post q/k/v conv. out = softmax(QK^T/sqrt(hd)) V,
// written as [B, C, N1]. 4 waves/block, 64 queries/block; K/V tiles (32 keys)
// are staged into LDS once per block and shared by all 4 waves.
constexpr int AP = 40;  // padded LDS stride (halves) for 32-wide attn tiles

__global__ __launch_bounds__(128) void attn_fused_kernel(
    const float* __restrict__ qf, const float* __restrict__ kf,
    const float* __restrict__ vf, float* __restrict__ outadd, int N1, int N2) {
  const int lane = threadIdx.x;
  const int wave = threadIdx.y;
  const int tid = wave * 32 + lane;
  const int m0 = blockIdx.x * 64 + wave * 16;  // this wave's query tile
  const int h = blockIdx.y;
  const int b = blockIdx.z;
  const float qscale = 0.17677669529663689f;  // 1/sqrt(32)

  const float* qh = qf + ((size_t)b * kC + h * kHD) * N1;
  const float* kh = kf + ((size_t)b * kC + h * kHD) * N2;
  const float* vh = vf + ((size_t)b * kC + h * kHD) * N2;
  float* oh = outadd + ((size_t)b * kC + h * kHD) * N1;

  __shared__ _Float16 Ks[32 * AP];  // [key][d]  (d contiguous -> score B frag)
  __shared__ _Float16 Vs[32 * AP];  // [d][key]  (key contiguous -> PV B frag)
  __shared__ float sS[4][16 * 32];
  __shared__ _Float16 sP[4][16 * 32];
  __shared__ float sfac[4][16], sinv[4][16];

  const int mb = (lane < 16) ? 0 : 8;
  const int nn = lane & 15;
  const int kbB = (lane < 16) ? 0 : 8;

  // Q fragment: A[m, k=d] = qh[d*N1 + m0+m]  (loaded once)
  v16h aq;
  {
    int m = m0 + nn;
#pragma unroll
    for (int e = 0; e < 16; ++e)
      aq[e] = (_Float16)qh[(size_t)a_k_of_e(lane, e) * N1 + m];
  }

  v8f acc0, acc1;
#pragma unroll
  for (int r = 0; r < 8; ++r) { acc0[r] = 0.0f; acc1[r] = 0.0f; }
  float rmax = -3.0e38f, rsum = 0.0f;  // rows owned by lanes 0..15

  for (int n2 = 0; n2 < N2; n2 += 32) {
    __syncthreads();  // previous iteration's Ks/Vs reads done
    // ---- stage K/V tiles (32 d x 32 keys) with coalesced float4 loads ----
#pragma unroll
    for (int rep = 0; rep < 2; ++rep) {
      int idx = rep * 128 + tid;  // 0..255 quads: 32 d-rows x 8 key-quads
      int d = idx >> 3, key4 = (idx & 7) * 4;
      v4f kv = *(const v4f*)(kh + (size_t)d * N2 + n2 + key4);
      v4f vv = *(const v4f*)(vh + (size_t)d * N2 + n2 + key4);
#pragma unroll
      for (int j = 0; j < 4; ++j) {
        Ks[(key4 + j) * AP + d] = (_Float16)kv[j];
        Vs[d * AP + key4 + j] = (_Float16)vv[j];
      }
    }
    if (n2 + 32 < N2) {  // prefetch next tile (global_prefetch_b8)
      __builtin_prefetch(kh + (size_t)(tid >> 2) * N2 + n2 + 32, 0, 1);
      __builtin_prefetch(vh + (size_t)(tid >> 2) * N2 + n2 + 32, 0, 1);
    }
    __syncthreads();

    // ---- scores: two 16x16 WMMA tiles; B frag = 2x ds_load_b128 ----
#pragma unroll
    for (int half = 0; half < 2; ++half) {
      const _Float16* kp = Ks + (size_t)(half * 16 + nn) * AP + kbB;
      v16h bk = combine16(*(const h8*)(kp), *(const h8*)(kp + 16));
      v8f z;
#pragma unroll
      for (int r = 0; r < 8; ++r) z[r] = 0.0f;
      v8f s = __builtin_amdgcn_wmma_f32_16x16x32_f16(false, aq, false, bk,
                                                     (short)0, z, false, false);
#pragma unroll
      for (int r = 0; r < 8; ++r)
        sS[wave][(mb + r) * 32 + half * 16 + nn] = s[r] * qscale;
    }
    __syncthreads();

    // ---- online softmax (lane m owns row m of this wave's tile) ----
    if (lane < 16) {
      const int m = lane;
      float tm = rmax;
      for (int j = 0; j < 32; ++j) tm = fmaxf(tm, sS[wave][m * 32 + j]);
      float fac = __expf(rmax - tm);
      float ps = 0.0f;
      for (int j = 0; j < 32; ++j) {
        float pv = __expf(sS[wave][m * 32 + j] - tm);
        sP[wave][m * 32 + j] = (_Float16)pv;
        ps += pv;
      }
      rsum = rsum * fac + ps;
      rmax = tm;
      sfac[wave][m] = fac;
    }
    __syncthreads();

#pragma unroll
    for (int r = 0; r < 8; ++r) {
      float f = sfac[wave][mb + r];
      acc0[r] *= f;
      acc1[r] *= f;
    }

    // ---- out += P x V; V frags = 2x ds_load_b128 each ----
    v16h ap;
#pragma unroll
    for (int e = 0; e < 16; ++e) ap[e] = sP[wave][nn * 32 + a_k_of_e(lane, e)];
    const _Float16* vp0 = Vs + (size_t)nn * AP + kbB;
    const _Float16* vp1 = Vs + (size_t)(16 + nn) * AP + kbB;
    v16h bv0 = combine16(*(const h8*)(vp0), *(const h8*)(vp0 + 16));
    v16h bv1 = combine16(*(const h8*)(vp1), *(const h8*)(vp1 + 16));
    acc0 = __builtin_amdgcn_wmma_f32_16x16x32_f16(false, ap, false, bv0,
                                                  (short)0, acc0, false, false);
    acc1 = __builtin_amdgcn_wmma_f32_16x16x32_f16(false, ap, false, bv1,
                                                  (short)0, acc1, false, false);
  }

  if (lane < 16) sinv[wave][lane] = 1.0f / rsum;
  __syncthreads();
#pragma unroll
  for (int r = 0; r < 8; ++r) {
    int m = mb + r;
    oh[(size_t)nn * N1 + (m0 + m)] = acc0[r] * sinv[wave][m];
    oh[(size_t)(16 + nn) * N1 + (m0 + m)] = acc1[r] * sinv[wave][m];
  }
}

// ---------------- Elementwise / reduction kernels --------------------------
__global__ void instnorm_bn_relu_kernel(const float* __restrict__ x,
                                        const float* __restrict__ g,
                                        const float* __restrict__ bp,
                                        float* __restrict__ y, int Np, int Cch) {
  const int row = blockIdx.x;  // b*C + c
  const int c = row % Cch;
  const float* xr = x + (size_t)row * Np;
  float* yr = y + (size_t)row * Np;
  __shared__ float r1[256], r2[256];
  const int t = threadIdx.x, T = blockDim.x;
  float s = 0.f, s2 = 0.f;
  for (int i = t; i < Np; i += T) { float v = xr[i]; s += v; s2 += v * v; }
  r1[t] = s; r2[t] = s2;
  __syncthreads();
  for (int k = T / 2; k > 0; k >>= 1) {
    if (t < k) { r1[t] += r1[t + k]; r2[t] += r2[t + k]; }
    __syncthreads();
  }
  const float mean = r1[0] / (float)Np;
  const float var = r2[0] / (float)Np - mean * mean;
  const float sc = g[c] * rsqrtf(1.f + 1e-5f) * rsqrtf(var + 1e-3f);
  const float bi = bp[c] - mean * sc;
  for (int i = t; i < Np; i += T) yr[i] = fmaxf(fmaf(xr[i], sc, bi), 0.f);
}

__global__ void softmax_rows_kernel(float* __restrict__ x, int n) {
  float* row = x + (size_t)blockIdx.x * n;
  __shared__ float red[256];
  const int t = threadIdx.x, T = blockDim.x;
  float m = -3.0e38f;
  for (int i = t; i < n; i += T) m = fmaxf(m, row[i]);
  red[t] = m;
  __syncthreads();
  for (int s = T / 2; s > 0; s >>= 1) {
    if (t < s) red[t] = fmaxf(red[t], red[t + s]);
    __syncthreads();
  }
  m = red[0];
  __syncthreads();
  float sum = 0.f;
  for (int i = t; i < n; i += T) {
    float e = __expf(row[i] - m);
    row[i] = e;
    sum += e;
  }
  red[t] = sum;
  __syncthreads();
  for (int s = T / 2; s > 0; s >>= 1) {
    if (t < s) red[t] += red[t + s];
    __syncthreads();
  }
  const float inv = 1.0f / red[0];
  for (int i = t; i < n; i += T) row[i] *= inv;
}

__global__ void bn_relu_kernel(float* __restrict__ x, const float* __restrict__ g,
                               const float* __restrict__ b, int Cch, int Np,
                               size_t total) {
  size_t i = (size_t)blockIdx.x * blockDim.x + threadIdx.x;
  if (i >= total) return;
  int c = (int)((i / (size_t)Np) % (size_t)Cch);
  x[i] = fmaxf(fmaf(x[i], g[c] * rsqrtf(1.f + 1e-5f), b[c]), 0.f);
}

__global__ void add_kernel(const float* __restrict__ a, const float* __restrict__ h,
                           float* __restrict__ d, size_t total) {
  size_t i = (size_t)blockIdx.x * blockDim.x + threadIdx.x;
  if (i < total) d[i] = a[i] + h[i];
}

__global__ void feats_weight_kernel(const float* __restrict__ fr,
                                    const float* __restrict__ g,
                                    const float* __restrict__ b,
                                    const float* __restrict__ fww,
                                    const float* __restrict__ fwbias,
                                    float* __restrict__ Wf) {
  int i = blockIdx.x * blockDim.x + threadIdx.x;
  if (i >= kB * kMS) return;
  const int bb = i / kMS, m = i % kMS;
  const float* frb = fr + (size_t)bb * kC * kMS + m;
  const float inv = rsqrtf(1.f + 1e-5f);
  float acc = fwbias[0];
  for (int c = 0; c < kC; ++c) {
    float v = fmaxf(fmaf(frb[(size_t)c * kMS], g[c] * inv, b[c]), 0.f);
    acc += fww[c] * v;
  }
  Wf[i] = 0.05f + 0.9f / (1.0f + __expf(-acc));
}

__global__ void build_kernel_mat_kernel(const float* __restrict__ G,
                                        const float* __restrict__ beta_p,
                                        float* __restrict__ Km) {
  int idx = blockIdx.x * blockDim.x + threadIdx.x;
  if (idx >= kB * kMS * kMS) return;
  const int bb = idx / (kMS * kMS);
  const int r = idx % (kMS * kMS);
  const int m = r / kMS, n = r % kMS;
  const float* Gb = G + (size_t)bb * kMS * kMS;
  const float beta = log1pf(__expf(beta_p[0]));  // softplus
  float d2 = fmaxf(Gb[(size_t)m * kMS + m] + Gb[(size_t)n * kMS + n] -
                       2.0f * Gb[r], 0.0f);
  Km[idx] = __expf(-beta * d2);
}

__global__ void build_A_kernel(const float* __restrict__ Km,
                               const float* __restrict__ Wf,
                               const float* __restrict__ lam_p,
                               float* __restrict__ Aout) {
  int idx = blockIdx.x * blockDim.x + threadIdx.x;
  if (idx >= kB * kMS * kMS) return;
  const int bb = idx / (kMS * kMS);
  const int r = idx % (kMS * kMS);
  const int m = r / kMS, n = r % kMS;
  const float lam = log1pf(__expf(lam_p[0])) + 2e-6f;  // softplus + 1e-6 + 1e-6
  float v = Km[idx] * Wf[bb * kMS + m] * Wf[bb * kMS + n];
  if (m == n) v += lam;
  Aout[idx] = v;
}

__global__ void build_wF_kernel(const float* __restrict__ val,
                                const float* __restrict__ Wf,
                                float* __restrict__ X) {
  int idx = blockIdx.x * blockDim.x + threadIdx.x;
  if (idx >= kB * kMS * kC) return;
  const int bb = idx / (kMS * kC);
  const int r = idx % (kMS * kC);
  const int m = r / kC, c = r % kC;
  X[idx] = val[(size_t)bb * kC * kMS + (size_t)c * kMS + m] * Wf[bb * kMS + m];
}

// In-place Gaussian elimination + back substitution, SPD A (512x512), 128 RHS.
__global__ void solve_kernel(float* __restrict__ A, float* __restrict__ X) {
  A += (size_t)blockIdx.x * kMS * kMS;
  X += (size_t)blockIdx.x * kMS * kC;
  const int t = threadIdx.x, T = blockDim.x;
  for (int k = 0; k < kMS; ++k) {
    __syncthreads();
    const float inv = 1.0f / A[(size_t)k * kMS + k];
    for (int i = k + 1 + t; i < kMS; i += T) {
      const float f = A[(size_t)i * kMS + k] * inv;
      for (int j = k + 1; j < kMS; ++j)
        A[(size_t)i * kMS + j] -= f * A[(size_t)k * kMS + j];
      for (int c = 0; c < kC; ++c)
        X[(size_t)i * kC + c] -= f * X[(size_t)k * kC + c];
    }
  }
  for (int k = kMS - 1; k >= 0; --k) {
    __syncthreads();
    const float inv = 1.0f / A[(size_t)k * kMS + k];
    for (int c = t; c < kC; c += T) X[(size_t)k * kC + c] *= inv;
    __syncthreads();
    const int tot = k * kC;
    for (int idx = t; idx < tot; idx += T) {
      const int i = idx / kC, c = idx % kC;
      X[(size_t)i * kC + c] -= A[(size_t)i * kMS + k] * X[(size_t)k * kC + c];
    }
  }
}

// ---------------- Host-side orchestration ----------------------------------
static inline size_t nblk(size_t total) { return (total + 255) / 256; }

// One AttentionPropagation block; pbase indexes 14 param arrays:
// wq,bq,wk,bk,wv,bv,wm,bm,wc1,bc1,g1,be1,wc2,bc2
static void attn_prop(hipStream_t st, const float* m1, int N1, const float* m2,
                      int N2, void* const* d_in, int pbase, float* dst,
                      float* qbuf, float* kbuf, float* vbuf, float* abuf,
                      float* mbuf, float* hbuf) {
  const float* wq = (const float*)d_in[pbase + 0];
  const float* bq = (const float*)d_in[pbase + 1];
  const float* wk = (const float*)d_in[pbase + 2];
  const float* bk = (const float*)d_in[pbase + 3];
  const float* wv = (const float*)d_in[pbase + 4];
  const float* bv = (const float*)d_in[pbase + 5];
  const float* wm = (const float*)d_in[pbase + 6];
  const float* bm = (const float*)d_in[pbase + 7];
  const float* wc1 = (const float*)d_in[pbase + 8];
  const float* bc1 = (const float*)d_in[pbase + 9];
  const float* g1 = (const float*)d_in[pbase + 10];
  const float* be1 = (const float*)d_in[pbase + 11];
  const float* wc2 = (const float*)d_in[pbase + 12];
  const float* bc2 = (const float*)d_in[pbase + 13];

  const size_t s1 = (size_t)kC * N1, s2 = (size_t)kC * N2;
  const size_t sh = (size_t)2 * kC * N1;

  launch_gemm(st, wq, kC, 0, m1, N1, s1, qbuf, N1, s1, bq, kC, N1, kC, kB, 0, 0, 0, 0);
  launch_gemm(st, wk, kC, 0, m2, N2, s2, kbuf, N2, s2, bk, kC, N2, kC, kB, 0, 0, 0, 0);
  launch_gemm(st, wv, kC, 0, m2, N2, s2, vbuf, N2, s2, bv, kC, N2, kC, kB, 0, 0, 0, 0);
  attn_fused_kernel<<<dim3(N1 / 64, kH, kB), dim3(32, 4), 0, st>>>(qbuf, kbuf, vbuf, abuf, N1, N2);
  launch_gemm(st, wm, kC, 0, abuf, N1, s1, mbuf, N1, s1, bm, kC, N1, kC, kB, 0, 0, 0, 0);
  // h = wc1[:, :C] x m1 + wc1[:, C:] x add + bc1   (avoids concat buffer)
  launch_gemm(st, wc1, 2 * kC, 0, m1, N1, s1, hbuf, N1, sh, bc1, 2 * kC, N1, kC, kB, 0, 0, 0, 0);
  launch_gemm(st, wc1 + kC, 2 * kC, 0, mbuf, N1, s1, hbuf, N1, sh, nullptr, 2 * kC, N1, kC, kB, 0, 0, 0, 1);
  {
    size_t tot = (size_t)kB * 2 * kC * N1;
    bn_relu_kernel<<<nblk(tot), 256, 0, st>>>(hbuf, g1, be1, 2 * kC, N1, tot);
  }
  launch_gemm(st, wc2, 2 * kC, 0, hbuf, N1, sh, abuf, N1, s1, bc2, kC, N1, 2 * kC, kB, 0, 0, 0, 0);
  {
    size_t tot = (size_t)kB * kC * N1;
    add_kernel<<<nblk(tot), 256, 0, st>>>(m1, abuf, dst, tot);
  }
}

extern "C" void kernel_launch(void* const* d_in, const int* in_sizes, int n_in,
                              void* d_out, int out_size, void* d_ws,
                              size_t ws_size, hipStream_t stream) {
  (void)in_sizes; (void)n_in; (void)out_size; (void)ws_size;

  const float* pos = (const float*)d_in[0];  // [B,C,N]
  const float* cf = (const float*)d_in[1];   // [B,C,N]
  const float* samp_g = (const float*)d_in[2];
  const float* samp_b = (const float*)d_in[3];
  const float* samp_wc = (const float*)d_in[4];  // [MS,C]
  const float* samp_bc = (const float*)d_in[5];  // [MS]
  // 6..19 inject, 20..33 rectify1, 34..47 rectify
  const float* fwg = (const float*)d_in[48];
  const float* fwb = (const float*)d_in[49];
  const float* fww = (const float*)d_in[50];
  const float* fwbias = (const float*)d_in[51];
  const float* kpw = (const float*)d_in[52];  // [C/2,C]
  const float* kpb = (const float*)d_in[53];
  const float* kvw = (const float*)d_in[54];  // [C,C]
  const float* kvb = (const float*)d_in[55];
  const float* beta_p = (const float*)d_in[56];
  const float* lam_p = (const float*)d_in[57];

  float* out = (float*)d_out;  // [B,C,N]

  // ---- workspace arena (~140 MB of f32) ----
  float* w = (float*)d_ws;
  float* xn = w;            w += (size_t)kB * kC * kN;   // reused as qbuf
  float* Wbuf = w;          w += (size_t)kB * kMS * kN;  // reused below
  float* hbuf = Wbuf;                                    // [B,2C,N]
  float* abuf = Wbuf + (size_t)kB * 2 * kC * kN;         // [B,C,N]
  float* mbuf = abuf + (size_t)kB * kC * kN;             // [B,C,N]
  float* kbuf = w;          w += (size_t)kB * kC * kN;
  float* vbuf = w;          w += (size_t)kB * kC * kN;
  float* fr = w;            w += (size_t)kB * kC * kMS;
  float* pM = w;            w += (size_t)kB * kC * kMS;
  float* fr2 = w;           w += (size_t)kB * kC * kMS;
  float* pre = w;           w += (size_t)kB * kC * kMS;
  float* posf = w;          w += (size_t)kB * (kC / 2) * kMS;
  float* val = w;           w += (size_t)kB * kC * kMS;
  float* Gm = w;            w += (size_t)kB * kMS * kMS;
  float* Km = w;            w += (size_t)kB * kMS * kMS;
  float* Am = w;            w += (size_t)kB * kMS * kMS;
  float* Xb = w;            w += (size_t)kB * kMS * kC;
  float* Wfb = w;           w += (size_t)kB * kMS;
  float* qbuf = xn;  // xn dead once W-logits are computed

  const size_t sCN = (size_t)kC * kN;
  const size_t sMSN = (size_t)kMS * kN;
  const size_t sCMS = (size_t)kC * kMS;

  // ---- AdaptiveSampling ----
  instnorm_bn_relu_kernel<<<kB * kC, 256, 0, stream>>>(cf, samp_g, samp_b, xn, kN, kC);
  launch_gemm(stream, samp_wc, kC, 0, xn, kN, sCN, Wbuf, kN, sMSN, samp_bc,
              kMS, kN, kC, kB, 0, 0, 0, 0);
  softmax_rows_kernel<<<kB * kMS, 256, 0, stream>>>(Wbuf, kN);
  // fr[c,m] = sum_n cf[c,n] * W[m,n]   (NT GEMM, K = 8192)
  launch_gemm(stream, cf, kN, sCN, Wbuf, kN, sMSN, fr, kMS, sCMS, nullptr,
              kC, kMS, kN, kB, 0, 1, 0, 0);
  launch_gemm(stream, pos, kN, sCN, Wbuf, kN, sMSN, pM, kMS, sCMS, nullptr,
              kC, kMS, kN, kB, 0, 1, 0, 0);

  // ---- inject: fr = attn_prop(fr, cf) ----
  attn_prop(stream, fr, kMS, cf, kN, d_in, 6, fr, qbuf, kbuf, vbuf, abuf, mbuf, hbuf);

  // ---- feats_weight -> Wf in [0.05, 0.95] ----
  feats_weight_kernel<<<nblk((size_t)kB * kMS), 256, 0, stream>>>(fr, fwg, fwb, fww, fwbias, Wfb);

  // ---- LearnableKernel ----
  launch_gemm(stream, kpw, kC, 0, pM, kMS, sCMS, posf, kMS, (size_t)(kC / 2) * kMS,
              kpb, kC / 2, kMS, kC, kB, 0, 0, 0, 0);
  launch_gemm(stream, kvw, kC, 0, fr, kMS, sCMS, val, kMS, sCMS, kvb, kC, kMS,
              kC, kB, 0, 0, 0, 0);
  // Gram G[m,n] = sum_d posf[d,m] posf[d,n]  (TN GEMM, K = 64)
  launch_gemm(stream, posf, kMS, (size_t)(kC / 2) * kMS, posf, kMS,
              (size_t)(kC / 2) * kMS, Gm, kMS, (size_t)kMS * kMS, nullptr,
              kMS, kMS, kC / 2, kB, 1, 0, 0, 0);
  build_kernel_mat_kernel<<<nblk((size_t)kB * kMS * kMS), 256, 0, stream>>>(Gm, beta_p, Km);
  build_A_kernel<<<nblk((size_t)kB * kMS * kMS), 256, 0, stream>>>(Km, Wfb, lam_p, Am);
  build_wF_kernel<<<nblk((size_t)kB * kMS * kC), 256, 0, stream>>>(val, Wfb, Xb);
  solve_kernel<<<kB, 256, 0, stream>>>(Am, Xb);
  // pre[c,m] = (kernel x Csol)[m,c]  (transposed store)
  launch_gemm(stream, Km, kMS, (size_t)kMS * kMS, Xb, kC, (size_t)kMS * kC,
              pre, kMS, sCMS, nullptr, kMS, kC, kMS, kB, 0, 0, 1, 0);

  // ---- rectify1: fr2 = attn_prop(fr, pre) ----
  attn_prop(stream, fr, kMS, pre, kMS, d_in, 20, fr2, qbuf, kbuf, vbuf, abuf, mbuf, hbuf);

  // ---- rectify: out = attn_prop(cf, fr2) ----
  attn_prop(stream, cf, kN, fr2, kMS, d_in, 34, out, qbuf, kbuf, vbuf, abuf, mbuf, hbuf);
}
